// NodeCountingLayer_36859409334286
// MI455X (gfx1250) — matrix operand order repaired and checked by reading, not compiled, not run on hardware
//
#include <hip/hip_runtime.h>
#include <hip/hip_bf16.h>

#define N_SAMPLES 1024
#define IN_FEAT   256
#define OUT_FEAT  256
#define N_OPS     4

typedef __attribute__((ext_vector_type(16))) __bf16 v16bf;
typedef __attribute__((ext_vector_type(8)))  float  v8f;

// 32-byte fragment: two uint4 chunks reinterpreted as 16 bf16 lanes-worth.
union Frag32B {
    struct { uint4 lo, hi; } q;
    v16bf v;
};

static __device__ inline unsigned short f2bf(float f) {
    // round-to-nearest-even float32 -> bf16
    unsigned u = __float_as_uint(f);
    u += 0x7FFFu + ((u >> 16) & 1u);
    return (unsigned short)(u >> 16);
}

// ---------------------------------------------------------------------------
// Prep 1: per-output-column straight-through selection code k*[o]
// ---------------------------------------------------------------------------
__global__ void prep_sel_kernel(const float* __restrict__ logits,
                                const float* __restrict__ gumbel,
                                int* __restrict__ codes) {
    int o = threadIdx.x;  // 256 threads, 1 block
    float best = logits[o * N_OPS] + gumbel[o * N_OPS];
    int bk = 0;
    #pragma unroll
    for (int k = 1; k < N_OPS; ++k) {
        float s = logits[o * N_OPS + k] + gumbel[o * N_OPS + k];
        if (s > best) { best = s; bk = k; }
    }
    codes[o] = bk;
}

// ---------------------------------------------------------------------------
// Prep 2: L0 = log(x), L1 = log(1-x) in bf16 (clamped so masked-out terms
// never produce inf*0 = NaN in the GEMM)
// ---------------------------------------------------------------------------
__global__ void prep_logs_kernel(const float* __restrict__ x,
                                 unsigned short* __restrict__ L0,
                                 unsigned short* __restrict__ L1) {
    int idx = blockIdx.x * blockDim.x + threadIdx.x;  // N_SAMPLES*IN_FEAT
    float xv = x[idx];
    L0[idx] = f2bf(__logf(fmaxf(xv, 1e-12f)));
    L1[idx] = f2bf(__logf(fmaxf(1.0f - xv, 1e-12f)));
}

// ---------------------------------------------------------------------------
// Prep 3: 0/1 masks per edge type (exact in bf16)
// ---------------------------------------------------------------------------
__global__ void prep_masks_kernel(const int* __restrict__ et,
                                  unsigned short* __restrict__ M0,
                                  unsigned short* __restrict__ M1) {
    int idx = blockIdx.x * blockDim.x + threadIdx.x;  // OUT_FEAT*IN_FEAT
    int e = et[idx];
    M0[idx] = (e == 0) ? (unsigned short)0x3F80 : (unsigned short)0;  // 1.0bf16
    M1[idx] = (e == 1) ? (unsigned short)0x3F80 : (unsigned short)0;
}

// ---------------------------------------------------------------------------
// Main GEMM: P = L0*M0^T + L1*M1^T, Q = L0*M1^T + L1*M0^T
// One wave32 per 16x16 output tile; K-loop of 8 x (4 WMMAs).
// A-frag layout (16-bit 16x32): lanes 0-15 row M, elems 0..7=K..K+7,
// 8..15=K+16..K+23; lanes 16-31 same rows, K shifted by 8.
// B-frag layout (16-bit 32x16): lanes 0-15 col N, elems 0..7=K..K+7,
// 8..15=K+16..K+23; lanes 16-31 same cols, K shifted by 8.
// ---------------------------------------------------------------------------
__global__ __launch_bounds__(256) void wmma_logprod_kernel(
        const unsigned short* __restrict__ L0,
        const unsigned short* __restrict__ L1,
        const unsigned short* __restrict__ M0,
        const unsigned short* __restrict__ M1,
        float* __restrict__ P, float* __restrict__ Q) {
    const int lane  = threadIdx.x & 31;
    const int wave  = (blockIdx.x * 256 + threadIdx.x) >> 5;  // 0..1023
    const int tile_m = (wave >> 4) << 4;   // 64 tiles of M
    const int tile_n = (wave & 15) << 4;   // 16 tiles of N
    const int lhalf  = lane >> 4;
    const int l15    = lane & 15;

    const unsigned short* a0row = L0 + (tile_m + l15) * IN_FEAT;
    const unsigned short* a1row = L1 + (tile_m + l15) * IN_FEAT;
    const unsigned short* b0row = M0 + (tile_n + l15) * IN_FEAT;
    const unsigned short* b1row = M1 + (tile_n + l15) * IN_FEAT;

    v8f accP = {};
    v8f accQ = {};

    for (int kb = 0; kb < IN_FEAT; kb += 32) {
        Frag32B fa0, fa1, fb0, fb1;
        const int aoff = kb + lhalf * 8;
        fa0.q.lo = *(const uint4*)(a0row + aoff);
        fa0.q.hi = *(const uint4*)(a0row + aoff + 16);
        fa1.q.lo = *(const uint4*)(a1row + aoff);
        fa1.q.hi = *(const uint4*)(a1row + aoff + 16);
        const int boff = kb + lhalf * 8;
        fb0.q.lo = *(const uint4*)(b0row + boff);
        fb0.q.hi = *(const uint4*)(b0row + boff + 16);
        fb1.q.lo = *(const uint4*)(b1row + boff);
        fb1.q.hi = *(const uint4*)(b1row + boff + 16);

        accP = __builtin_amdgcn_wmma_f32_16x16x32_bf16(
                   false, fa0.v, false, fb0.v, (short)0, accP, false, false);
        accP = __builtin_amdgcn_wmma_f32_16x16x32_bf16(
                   false, fa1.v, false, fb1.v, (short)0, accP, false, false);
        accQ = __builtin_amdgcn_wmma_f32_16x16x32_bf16(
                   false, fa0.v, false, fb1.v, (short)0, accQ, false, false);
        accQ = __builtin_amdgcn_wmma_f32_16x16x32_bf16(
                   false, fa1.v, false, fb0.v, (short)0, accQ, false, false);
    }

    // C/D layout: VGPR r -> row (lhalf*8 + r), col l15
    const int mbase = tile_m + lhalf * 8;
    const int ncol  = tile_n + l15;
    #pragma unroll
    for (int r = 0; r < 8; ++r) {
        P[(mbase + r) * OUT_FEAT + ncol] = accP[r];
        Q[(mbase + r) * OUT_FEAT + ncol] = accQ[r];
    }
}

// ---------------------------------------------------------------------------
// Combine: pick selected op per column; min/max columns do a VALU reduction.
// ---------------------------------------------------------------------------
__global__ void combine_kernel(const float* __restrict__ x,
                               const int* __restrict__ et,
                               const int* __restrict__ codes,
                               const float* __restrict__ P,
                               const float* __restrict__ Q,
                               float* __restrict__ out) {
    int idx = blockIdx.x * blockDim.x + threadIdx.x;  // n*OUT_FEAT + o
    int o = idx & (OUT_FEAT - 1);
    int n = idx >> 8;
    int code = codes[o];
    float r;
    if (code == 1) {
        r = __expf(P[idx]);                 // op_prod
    } else if (code == 3) {
        r = 1.0f - __expf(Q[idx]);          // op_psum
    } else {
        const float* xr = x + n * IN_FEAT;
        const int*   er = et + o * IN_FEAT;
        if (code == 0) {                    // op_min over v_tnorm
            r = 1.0f;
            for (int i = 0; i < IN_FEAT; ++i) {
                int e = er[i];
                float v = (e == 2) ? 1.0f : ((e == 1) ? 1.0f - xr[i] : xr[i]);
                r = fminf(r, v);
            }
        } else {                            // op_max over v_tconorm
            r = 0.0f;
            for (int i = 0; i < IN_FEAT; ++i) {
                int e = er[i];
                float v = (e == 2) ? 0.0f : ((e == 1) ? 1.0f - xr[i] : xr[i]);
                r = fmaxf(r, v);
            }
        }
    }
    out[idx] = r;
}

// ---------------------------------------------------------------------------
// Fallback (only if workspace is too small): fully fused naive kernel.
// ---------------------------------------------------------------------------
__global__ void naive_all_kernel(const float* __restrict__ x,
                                 const float* __restrict__ logits,
                                 const float* __restrict__ gumbel,
                                 const int* __restrict__ et,
                                 float* __restrict__ out) {
    int idx = blockIdx.x * blockDim.x + threadIdx.x;
    int o = idx & (OUT_FEAT - 1);
    int n = idx >> 8;
    float best = logits[o * N_OPS] + gumbel[o * N_OPS];
    int bk = 0;
    #pragma unroll
    for (int k = 1; k < N_OPS; ++k) {
        float s = logits[o * N_OPS + k] + gumbel[o * N_OPS + k];
        if (s > best) { best = s; bk = k; }
    }
    const float* xr = x + n * IN_FEAT;
    const int*   er = et + o * IN_FEAT;
    float r;
    if (bk == 0) {
        r = 1.0f;
        for (int i = 0; i < IN_FEAT; ++i) {
            int e = er[i];
            float v = (e == 2) ? 1.0f : ((e == 1) ? 1.0f - xr[i] : xr[i]);
            r = fminf(r, v);
        }
    } else if (bk == 1) {
        r = 1.0f;
        for (int i = 0; i < IN_FEAT; ++i) {
            int e = er[i];
            float v = (e == 2) ? 1.0f : ((e == 1) ? 1.0f - xr[i] : xr[i]);
            r *= v;
        }
    } else if (bk == 2) {
        r = 0.0f;
        for (int i = 0; i < IN_FEAT; ++i) {
            int e = er[i];
            float v = (e == 2) ? 0.0f : ((e == 1) ? 1.0f - xr[i] : xr[i]);
            r = fmaxf(r, v);
        }
    } else {
        float p = 1.0f;
        for (int i = 0; i < IN_FEAT; ++i) {
            int e = er[i];
            float v = (e == 2) ? 0.0f : ((e == 1) ? 1.0f - xr[i] : xr[i]);
            p *= (1.0f - v);
        }
        r = 1.0f - p;
    }
    out[idx] = r;
}

extern "C" void kernel_launch(void* const* d_in, const int* in_sizes, int n_in,
                              void* d_out, int out_size, void* d_ws, size_t ws_size,
                              hipStream_t stream) {
    const float* x      = (const float*)d_in[0];
    const float* logits = (const float*)d_in[1];
    const float* gumbel = (const float*)d_in[2];
    const int*   et     = (const int*)d_in[3];
    float* out = (float*)d_out;

    // Workspace layout (all offsets 16B-aligned)
    const size_t offL0 = 0;
    const size_t offL1 = offL0 + (size_t)N_SAMPLES * IN_FEAT * 2;
    const size_t offM0 = offL1 + (size_t)N_SAMPLES * IN_FEAT * 2;
    const size_t offM1 = offM0 + (size_t)OUT_FEAT * IN_FEAT * 2;
    const size_t offP  = offM1 + (size_t)OUT_FEAT * IN_FEAT * 2;
    const size_t offQ  = offP  + (size_t)N_SAMPLES * OUT_FEAT * 4;
    const size_t offC  = offQ  + (size_t)N_SAMPLES * OUT_FEAT * 4;
    const size_t need  = offC  + (size_t)OUT_FEAT * 4;

    if (ws_size >= need) {
        char* ws = (char*)d_ws;
        unsigned short* L0 = (unsigned short*)(ws + offL0);
        unsigned short* L1 = (unsigned short*)(ws + offL1);
        unsigned short* M0 = (unsigned short*)(ws + offM0);
        unsigned short* M1 = (unsigned short*)(ws + offM1);
        float* P   = (float*)(ws + offP);
        float* Q   = (float*)(ws + offQ);
        int* codes = (int*)(ws + offC);

        prep_sel_kernel<<<1, OUT_FEAT, 0, stream>>>(logits, gumbel, codes);
        prep_logs_kernel<<<(N_SAMPLES * IN_FEAT) / 256, 256, 0, stream>>>(x, L0, L1);
        prep_masks_kernel<<<(OUT_FEAT * IN_FEAT) / 256, 256, 0, stream>>>(et, M0, M1);
        // 1024 tiles = 1024 waves = 128 blocks x 8 waves
        wmma_logprod_kernel<<<128, 256, 0, stream>>>(L0, L1, M0, M1, P, Q);
        combine_kernel<<<(N_SAMPLES * OUT_FEAT) / 256, 256, 0, stream>>>(x, et, codes, P, Q, out);
    } else {
        naive_all_kernel<<<(N_SAMPLES * OUT_FEAT) / 256, 256, 0, stream>>>(
            x, logits, gumbel, et, out);
    }
}
